// SimpleSelfAttention_70437463654765
// MI455X (gfx1250) — matrix operand order repaired and checked by reading, not compiled, not run on hardware
//
#include <hip/hip_runtime.h>
#include <stdint.h>

// ---------------------------------------------------------------------------
// Causal self-attention, B=4 S=2048 D=1024, via CDNA5 WMMA (gfx1250, wave32).
//
// ~120 GFLOP vs ~80MB unique IO -> matrix-unit bound.  All three GEMMs run on
// v_wmma_f32_16x16x32_f16.  Block = 2x2 waves = 64x64 output tile; K staged
// in 128-wide double-buffered LDS tiles filled with GLOBAL_LOAD_ASYNC_TO_LDS
// (ASYNCcnt) so one copy feeds all 4 waves and overlaps the WMMA pipe.
// ---------------------------------------------------------------------------

typedef __attribute__((ext_vector_type(16))) _Float16 v16h;
typedef __attribute__((ext_vector_type(8)))  _Float16 v8h;
typedef __attribute__((ext_vector_type(8)))  float    v8f;

#define BB   4
#define SS   2048
#define DD   1024
#define NROW (BB * SS)          // 8192 flattened (b,s) rows
#define SCL  0.03125f           // 1/sqrt(1024)
#define BK   128                // k-chunk per LDS stage
#define BM   64                 // block tile rows
#define BN   64                 // block tile cols

// ---- workspace layout (bytes) ---------------------------------------------
#define OFF_XH 0u                       // x   f16  [8192,1024]   16 MB
#define OFF_WH 16777216u                // w   f16  [3,1024,1024]  6 MB
#define OFF_QH 23068672u                // Q   f16  [8192,1024]   16 MB
#define OFF_KH 39845888u                // K   f16  [8192,1024]   16 MB
#define OFF_VT 56623104u                // V^T f16  [4,1024,2048] 16 MB
#define OFF_SC 73400320u                // S   f32  [4,2048,2048] 64 MB
#define OFF_P  140509184u               // P   f16  [4,2048,2048] 32 MB

// ---------------------------------------------------------------------------
// WMMA fragment loaders (ISA 7.12.2, 16-bit A 16x32 / B 32x16, wave32).
// A: lane m=lane&15 holds K = base..base+7 and base+16..base+23, base=k0+8*hi.
// B: lane n=lane&15 holds K = base..base+7 and base+16..base+23, base=k0+8*hi.
// ---------------------------------------------------------------------------
__device__ __forceinline__ v16h frag_a(const _Float16* __restrict__ p, int ld,
                                       int row0, int k0) {
  int lane = threadIdx.x & 31;
  const _Float16* r =
      p + (size_t)(row0 + (lane & 15)) * ld + k0 + ((lane >> 4) << 3);
  v8h lo = *(const v8h*)(r);
  v8h hi = *(const v8h*)(r + 16);
  return __builtin_shufflevector(lo, hi, 0, 1, 2, 3, 4, 5, 6, 7, 8, 9, 10, 11,
                                 12, 13, 14, 15);
}

__device__ __forceinline__ v16h frag_b(const _Float16* __restrict__ p, int ld,
                                       int row0, int k0) {
  int lane = threadIdx.x & 31;
  const _Float16* r =
      p + (size_t)(row0 + (lane & 15)) * ld + k0 + ((lane >> 4) << 3);
  v8h lo = *(const v8h*)(r);
  v8h hi = *(const v8h*)(r + 16);
  return __builtin_shufflevector(lo, hi, 0, 1, 2, 3, 4, 5, 6, 7, 8, 9, 10, 11,
                                 12, 13, 14, 15);
}

__device__ __forceinline__ v8f wmma_f16(v16h a, v16h b, v8f c) {
  return __builtin_amdgcn_wmma_f32_16x16x32_f16(false, a, false, b, (short)0,
                                                c, false, false);
}

// C/D tile layout: VGPR r <-> row m0 + r + 8*(lane>=16), col n0 + (lane&15).
__device__ __forceinline__ void store_f32(float* __restrict__ out, int ld,
                                          int m0, int n0, v8f c) {
  int lane = threadIdx.x & 31;
  int col = n0 + (lane & 15);
  int rbase = m0 + ((lane >> 4) << 3);
  for (int r = 0; r < 8; ++r) out[(size_t)(rbase + r) * ld + col] = c[r];
}

__device__ __forceinline__ void store_f16(_Float16* __restrict__ out, int ld,
                                          int m0, int n0, v8f c) {
  int lane = threadIdx.x & 31;
  int col = n0 + (lane & 15);
  int rbase = m0 + ((lane >> 4) << 3);
  for (int r = 0; r < 8; ++r)
    out[(size_t)(rbase + r) * ld + col] = (_Float16)c[r];
}

// ---------------------------------------------------------------------------
// Async copy: 16 bytes global -> LDS, tracked by ASYNCcnt (ISA 10 / 15.18.3).
// GV addressing: LDS address comes from the VDST VGPR; generic LDS pointers carry
// the LDS byte offset in their low 32 bits (ISA 10.2 aperture rule).
// ---------------------------------------------------------------------------
__device__ __forceinline__ void async_cp16(uint32_t lds_off, const void* g) {
  asm volatile("global_load_async_to_lds_b128 %0, %1, off"
               :
               : "v"(lds_off), "v"((uint64_t)(uintptr_t)g)
               : "memory");
}

// ---------------------------------------------------------------------------
// Shared 64x64 block GEMM: C += A[arow0: ,k] . B[brow0: ,k]^T with both
// operands row-major [rows, ld] f16.  kmax is rounded up to BK internally
// (callers guarantee over-read stays inside the row and contributes zero).
// ---------------------------------------------------------------------------
__device__ __forceinline__ void gemm_block_64x64(
    const _Float16* __restrict__ A, int lda, int arow0,
    const _Float16* __restrict__ Bm, int ldb, int brow0, int kmax,
    v8f& c00, v8f& c01, v8f& c10, v8f& c11, int mloc, int nloc) {
  __shared__ _Float16 As[2][BM * BK];
  __shared__ _Float16 Bs[2][BM * BK];
  const int t = threadIdx.x;                 // 0..127
  const char* Abase = (const char*)(A + (size_t)arow0 * lda);
  const char* Bbase = (const char*)(Bm + (size_t)brow0 * ldb);
  const int nstages = (kmax + BK - 1) / BK;

  auto issue = [&](int s, int buf) {
    uint32_t la = (uint32_t)(uintptr_t)&As[buf][0];
    uint32_t lb = (uint32_t)(uintptr_t)&Bs[buf][0];
    int kbyte = s * (BK * 2);                // stage byte offset within row
    for (int i = 0; i < 8; ++i) {            // 1024 x 16B per tile, 128 thr
      int e = t + i * 128;                   // 0..1023
      int row = e >> 4;                      // 0..63
      int cb = (e & 15) * 16;                // byte within 256B row chunk
      uint32_t lo = (uint32_t)(row * (BK * 2) + cb);
      async_cp16(la + lo, Abase + (size_t)row * lda * 2 + kbyte + cb);
      async_cp16(lb + lo, Bbase + (size_t)row * ldb * 2 + kbyte + cb);
    }
  };

  issue(0, 0);
  for (int s = 0; s < nstages; ++s) {
    int buf = s & 1;
    if (s + 1 < nstages) {
      issue(s + 1, buf ^ 1);
      // 16 newer async ops in flight; <=16 left => stage s fully landed
      asm volatile("s_wait_asynccnt 16" ::: "memory");
    } else {
      asm volatile("s_wait_asynccnt 0" ::: "memory");
    }
    __syncthreads();
    for (int kk = 0; kk < BK; kk += 32) {
      v16h a0 = frag_a(&As[buf][0], BK, mloc, kk);
      v16h a1 = frag_a(&As[buf][0], BK, mloc + 16, kk);
      v16h b0 = frag_b(&Bs[buf][0], BK, nloc, kk);
      v16h b1 = frag_b(&Bs[buf][0], BK, nloc + 16, kk);
      c00 = wmma_f16(a0, b0, c00);
      c01 = wmma_f16(a0, b1, c01);
      c10 = wmma_f16(a1, b0, c10);
      c11 = wmma_f16(a1, b1, c11);
    }
    __syncthreads();                          // buf consumed by all waves
  }
}

// ---------------------------------------------------------------------------
// K0: f32 -> f16 elementwise conversion
// ---------------------------------------------------------------------------
__global__ void cvt_f32_f16(const float* __restrict__ in,
                            _Float16* __restrict__ out, int n) {
  int i = blockIdx.x * blockDim.x + threadIdx.x;
  if (i < n) out[i] = (_Float16)in[i];
}

// ---------------------------------------------------------------------------
// K1: fused QKV projection.  Block: 64x64 of [Q|K|V]; V written transposed
// [b, d, s] so the PV GEMM's B operand is contiguous.
// ---------------------------------------------------------------------------
__global__ void qkv_gemm(const _Float16* __restrict__ xh,
                         const _Float16* __restrict__ wh,
                         _Float16* __restrict__ qh, _Float16* __restrict__ kh,
                         _Float16* __restrict__ vt) {
  int wave = threadIdx.x >> 5;
  int mloc = (wave >> 1) * 32, nloc = (wave & 1) * 32;
  int m0 = blockIdx.x * BM;                  // 0..8128
  int ng = blockIdx.y * BN;                  // 0..3008 (64 | 1024)
  int sel = ng >> 10;                        // 0=Q 1=K 2=V
  int n0 = ng & (DD - 1);
  const _Float16* w = wh + (size_t)sel * DD * DD;

  v8f c00 = {}, c01 = {}, c10 = {}, c11 = {};
  gemm_block_64x64(xh, DD, m0, w, DD, n0, DD, c00, c01, c10, c11, mloc, nloc);

  int mw = m0 + mloc, nw = n0 + nloc;
  if (sel == 0 || sel == 1) {
    _Float16* dst = (sel == 0) ? qh : kh;
    store_f16(dst, DD, mw, nw, c00);
    store_f16(dst, DD, mw, nw + 16, c01);
    store_f16(dst, DD, mw + 16, nw, c10);
    store_f16(dst, DD, mw + 16, nw + 16, c11);
  } else {
    int lane = threadIdx.x & 31;
    v8f cs[4] = {c00, c01, c10, c11};
    for (int q = 0; q < 4; ++q) {
      int mo = mw + (q >> 1) * 16 + ((lane >> 4) << 3);
      int col = nw + (q & 1) * 16 + (lane & 15);
      for (int r = 0; r < 8; ++r) {
        int row = mo + r;
        int bi = row >> 11, si = row & (SS - 1);
        vt[((size_t)bi * DD + col) * SS + si] = (_Float16)cs[q][r];
      }
    }
  }
}

// ---------------------------------------------------------------------------
// K2: scores = Q . K^T over D=1024.  Block: 64q x 64k.  Blocks entirely
// above the causal diagonal exit before touching LDS (block-uniform branch).
// ---------------------------------------------------------------------------
__global__ void scores_gemm(const _Float16* __restrict__ qh,
                            const _Float16* __restrict__ kh,
                            float* __restrict__ sc) {
  int bi = blockIdx.z;
  int q0 = blockIdx.x * BM;
  int k0 = blockIdx.y * BN;
  if (k0 > q0 + (BM - 1)) return;            // fully masked block

  int wave = threadIdx.x >> 5;
  int mloc = (wave >> 1) * 32, nloc = (wave & 1) * 32;
  const _Float16* Q = qh + (size_t)bi * SS * DD;
  const _Float16* K = kh + (size_t)bi * SS * DD;

  v8f c00 = {}, c01 = {}, c10 = {}, c11 = {};
  gemm_block_64x64(Q, DD, q0, K, DD, k0, DD, c00, c01, c10, c11, mloc, nloc);

  float* out = sc + (size_t)bi * SS * SS;
  int mw = q0 + mloc, nw = k0 + nloc;
  store_f32(out, SS, mw, nw, c00);
  store_f32(out, SS, mw, nw + 16, c01);
  store_f32(out, SS, mw + 16, nw, c10);
  store_f32(out, SS, mw + 16, nw + 16, c11);
}

// ---------------------------------------------------------------------------
// K3: causal row softmax with scale, f32 scores -> f16 probabilities.
// Writes zeros above the diagonal so the PV GEMM needs no masking.
// ---------------------------------------------------------------------------
__global__ void softmax_rows(const float* __restrict__ sc,
                             _Float16* __restrict__ P) {
  __shared__ float red[256];
  int row = blockIdx.x;                      // 0..8191  (= b*S + q)
  int q = row & (SS - 1);
  int L = q + 1;
  const float* s = sc + (size_t)row * SS;
  _Float16* p = P + (size_t)row * SS;
  int t = threadIdx.x;

  float mx = -3.402823466e38f;
  for (int k = t; k < L; k += 256) mx = fmaxf(mx, s[k]);
  red[t] = mx;
  __syncthreads();
  for (int o = 128; o > 0; o >>= 1) {
    if (t < o) red[t] = fmaxf(red[t], red[t + o]);
    __syncthreads();
  }
  mx = red[0];
  __syncthreads();

  float sum = 0.f;
  for (int k = t; k < L; k += 256) sum += __expf(SCL * (s[k] - mx));
  red[t] = sum;
  __syncthreads();
  for (int o = 128; o > 0; o >>= 1) {
    if (t < o) red[t] += red[t + o];
    __syncthreads();
  }
  float inv = 1.0f / red[0];

  for (int k = t; k < SS; k += 256) {
    float v = (k < L) ? __expf(SCL * (s[k] - mx)) * inv : 0.f;
    p[k] = (_Float16)v;
  }
}

// ---------------------------------------------------------------------------
// K4: O = P . V, contraction clipped to causal extent (kmax = q0+64; the
// staging rounds up to 128 — over-read stays in the row and P is zero there).
// B operand reads V^T [b, d, s] row-major.
// ---------------------------------------------------------------------------
__global__ void pv_gemm(const _Float16* __restrict__ P,
                        const _Float16* __restrict__ vt,
                        float* __restrict__ out) {
  int wave = threadIdx.x >> 5;
  int mloc = (wave >> 1) * 32, nloc = (wave & 1) * 32;
  int m0 = blockIdx.x * BM;                  // flattened (b,s) row tile
  int n0 = blockIdx.y * BN;                  // d column tile
  int bi = m0 >> 11;
  int q0 = m0 & (SS - 1);
  const _Float16* Pb = P + (size_t)bi * SS * SS;
  const _Float16* Vb = vt + (size_t)bi * DD * SS;
  int kmax = q0 + BM;                        // <= SS

  v8f c00 = {}, c01 = {}, c10 = {}, c11 = {};
  gemm_block_64x64(Pb, SS, q0, Vb, SS, n0, kmax, c00, c01, c10, c11, mloc,
                   nloc);

  int mw = m0 + mloc, nw = n0 + nloc;
  store_f32(out, DD, mw, nw, c00);
  store_f32(out, DD, mw, nw + 16, c01);
  store_f32(out, DD, mw + 16, nw, c10);
  store_f32(out, DD, mw + 16, nw + 16, c11);
}

// ---------------------------------------------------------------------------
extern "C" void kernel_launch(void* const* d_in, const int* in_sizes, int n_in,
                              void* d_out, int out_size, void* d_ws,
                              size_t ws_size, hipStream_t stream) {
  const float* x = (const float*)d_in[0];
  const float* wq = (const float*)d_in[1];
  const float* wk = (const float*)d_in[2];
  const float* wv = (const float*)d_in[3];
  float* out = (float*)d_out;

  char* ws = (char*)d_ws;
  _Float16* xh = (_Float16*)(ws + OFF_XH);
  _Float16* wh = (_Float16*)(ws + OFF_WH);
  _Float16* qh = (_Float16*)(ws + OFF_QH);
  _Float16* kh = (_Float16*)(ws + OFF_KH);
  _Float16* vt = (_Float16*)(ws + OFF_VT);
  float*    sc = (float*)(ws + OFF_SC);
  _Float16* P  = (_Float16*)(ws + OFF_P);

  const int nx = NROW * DD;     // 8,388,608
  const int nw = DD * DD;       // 1,048,576

  cvt_f32_f16<<<(nx + 255) / 256, 256, 0, stream>>>(x, xh, nx);
  cvt_f32_f16<<<(nw + 255) / 256, 256, 0, stream>>>(wq, wh, nw);
  cvt_f32_f16<<<(nw + 255) / 256, 256, 0, stream>>>(wk, wh + nw, nw);
  cvt_f32_f16<<<(nw + 255) / 256, 256, 0, stream>>>(wv, wh + 2 * nw, nw);

  // 128 m-blocks x 48 n-blocks (Q|K|V), 64x64 per block of 4 waves
  qkv_gemm<<<dim3(NROW / BM, (3 * DD) / BN), 128, 0, stream>>>(xh, wh, qh, kh,
                                                               vt);

  // 32 q-blocks x 32 k-blocks x 4 batches
  scores_gemm<<<dim3(SS / BM, SS / BN, BB), 128, 0, stream>>>(qh, kh, sc);

  softmax_rows<<<NROW, 256, 0, stream>>>(sc, P);

  // 128 m-blocks x 16 n-blocks
  pv_gemm<<<dim3(NROW / BM, DD / BN), 128, 0, stream>>>(P, vt, out);
}
